// MambaRouter_51299089383766
// MI455X (gfx1250) — matrix operand order repaired and checked by reading, not compiled, not run on hardware
//
#include <hip/hip_runtime.h>
#include <hip/hip_bf16.h>
#include <math.h>

// MI455X / gfx1250, wave32. All GEMMs via v_wmma_f32_16x16x32_bf16.
typedef __attribute__((ext_vector_type(16))) __bf16 v16bf;
typedef __attribute__((ext_vector_type(8)))  __bf16 v8bf;
typedef __attribute__((ext_vector_type(8)))  float  v8f;

#define WAVES_PER_BLOCK 4
#define ROWS_PER_WAVE   16
#define TILE_B          (WAVES_PER_BLOCK * ROWS_PER_WAVE)   // 64 rows / block
#define INPUT_DIM 768
#define D_MODEL   64
#define D_INNER   128
#define D_STATE   16
#define DT_RANK   4
#define XPROJ_N   36            // dt(4) + B(16) + C(16)
#define XPROJ_NP  48            // padded to 3 N-tiles

// bf16 weight workspace layout (elements)
#define WS_PROJ   0                          // [64][768]
#define WS_MIN    (WS_PROJ + 64 * 768)       // [256][64]
#define WS_XPROJ  (WS_MIN + 256 * 64)        // [48][128] zero-padded
#define WS_OUT    (WS_XPROJ + XPROJ_NP * 128)// [64][128]
#define WS_TOTAL  (WS_OUT + 64 * 128)

static __device__ __forceinline__ v8f wmma_bf16(v16bf a, v16bf b, v8f c) {
    return __builtin_amdgcn_wmma_f32_16x16x32_bf16(
        false, a, false, b, (short)0, c, false, false);
}

// A-fragment (16x32, MxK) from f32 source (converted): lane L holds row M=L%16;
// lanes 0-15: K = kb+[0..7], kb+[16..23]; lanes 16-31: K = kb+[8..15], kb+[24..31].
static __device__ __forceinline__ v16bf load_a_frag_f32(const float* base, int ld, int kb, int lane) {
    const int M = lane & 15, g = lane >> 4;
    const float* p0 = base + (size_t)M * ld + kb + g * 8;
    const float* p1 = p0 + 16;
    v16bf a;
#pragma unroll
    for (int j = 0; j < 8; ++j) {
        a[j]     = (__bf16)p0[j];
        a[j + 8] = (__bf16)p1[j];
    }
    return a;
}

// A-fragment from bf16 source: two aligned 16B vector loads, no VALU conversion.
static __device__ __forceinline__ v16bf load_a_frag_bf(const __bf16* base, int ld, int kb, int lane) {
    const int M = lane & 15, g = lane >> 4;
    const v8bf lo = *(const v8bf*)(base + (size_t)M * ld + kb + g * 8);
    const v8bf hi = *(const v8bf*)(base + (size_t)M * ld + kb + 16 + g * 8);
    return __builtin_shufflevector(lo, hi, 0, 1, 2, 3, 4, 5, 6, 7,
                                           8, 9, 10, 11, 12, 13, 14, 15);
}

// B-fragment (32x16, KxN) from row-major bf16 W[N][K]: lane L holds col
// N = nb + L%16, K = kb + (L/16)*8 + [0..7], kb + 16 + (L/16)*8 + [0..7] -> two aligned 16B loads.
static __device__ __forceinline__ v16bf load_b_frag_bf(const __bf16* base, int ld, int nb, int kb, int lane) {
    const int N = nb + (lane & 15), g = lane >> 4;
    const v8bf lo = *(const v8bf*)(base + (size_t)N * ld + kb + g * 8);
    const v8bf hi = *(const v8bf*)(base + (size_t)N * ld + kb + 16 + g * 8);
    return __builtin_shufflevector(lo, hi, 0, 1, 2, 3, 4, 5, 6, 7,
                                           8, 9, 10, 11, 12, 13, 14, 15);
}

static __device__ __forceinline__ float silu_f(float v)     { return v / (1.f + __expf(-v)); }
static __device__ __forceinline__ float softplus_f(float v) { return (v > 20.f) ? v : log1pf(__expf(v)); }

// ---- one-shot weight conversion into d_ws (bf16, padded xproj) --------------
__global__ void pack_weights(const float* __restrict__ proj_W,
                             const float* __restrict__ min_W,
                             const float* __restrict__ xproj_W,
                             const float* __restrict__ out_W,
                             __bf16* __restrict__ ws) {
    const int tid = blockIdx.x * blockDim.x + threadIdx.x;
    const int stride = gridDim.x * blockDim.x;
    for (int i = tid; i < 64 * 768; i += stride)  ws[WS_PROJ + i] = (__bf16)proj_W[i];
    for (int i = tid; i < 256 * 64; i += stride)  ws[WS_MIN + i]  = (__bf16)min_W[i];
    for (int i = tid; i < XPROJ_NP * 128; i += stride) {
        const int e = i / 128, k = i % 128;
        ws[WS_XPROJ + i] = (__bf16)((e < XPROJ_N) ? xproj_W[e * 128 + k] : 0.f);
    }
    for (int i = tid; i < 64 * 128; i += stride)  ws[WS_OUT + i] = (__bf16)out_W[i];
}

__global__ __launch_bounds__(WAVES_PER_BLOCK * 32)
void mamba_router_fused(const float* __restrict__ x,        // [B,768]
                        const __bf16* __restrict__ ws,      // packed bf16 weights
                        const float* __restrict__ proj_b,   // [64]
                        const float* __restrict__ conv_W,   // [128,1,4]
                        const float* __restrict__ conv_b,   // [128]
                        const float* __restrict__ dt_W,     // [128,4]
                        const float* __restrict__ dt_b,     // [128]
                        const float* __restrict__ Dparam,   // [128]
                        const float* __restrict__ head_W,   // [2,64]
                        const float* __restrict__ head_b,   // [2]
                        float* __restrict__ out)            // [B,2]
{
    __shared__ __bf16 s_hb [WAVES_PER_BLOCK][ROWS_PER_WAVE * D_MODEL];
    __shared__ float  s_xc [WAVES_PER_BLOCK][ROWS_PER_WAVE * D_INNER];
    __shared__ float  s_dbc[WAVES_PER_BLOCK][ROWS_PER_WAVE * XPROJ_NP];
    __shared__ float  s_S  [WAVES_PER_BLOCK][ROWS_PER_WAVE];
    __shared__ __bf16 s_yb [WAVES_PER_BLOCK][ROWS_PER_WAVE * D_INNER];
    __shared__ float  s_m  [WAVES_PER_BLOCK][ROWS_PER_WAVE * D_MODEL];

    const __bf16* wproj = ws + WS_PROJ;
    const __bf16* wmin  = ws + WS_MIN;
    const __bf16* wxp   = ws + WS_XPROJ;
    const __bf16* wout  = ws + WS_OUT;

    const int tid  = threadIdx.x;
    const int wave = tid >> 5;
    const int lane = tid & 31;
    const int g    = lane >> 4;      // half-wave
    const int c    = lane & 15;      // column within tile
    const int rowBase = blockIdx.x * TILE_B + wave * ROWS_PER_WAVE;

    // ---------------- GEMM1: h[16x64] = x[16x768] @ proj_W^T -----------------
    v8f accH[4] = {};
    {
        const float* xrow = x + (size_t)rowBase * INPUT_DIM;
#pragma unroll 4
        for (int kk = 0; kk < INPUT_DIM; kk += 32) {
            v16bf af = load_a_frag_f32(xrow, INPUT_DIM, kk, lane);
            if (kk + 64 < INPUT_DIM)   // stream-prefetch next A chunk
                __builtin_prefetch(xrow + (size_t)(lane & 15) * INPUT_DIM + kk + 64, 0, 0);
#pragma unroll
            for (int n = 0; n < 4; ++n)
                accH[n] = wmma_bf16(af, load_b_frag_bf(wproj, INPUT_DIM, n * 16, kk, lane), accH[n]);
        }
    }
    // C-layout -> row-major bf16 in LDS (+ bias) for the next GEMM's A feed.
#pragma unroll
    for (int n = 0; n < 4; ++n)
#pragma unroll
        for (int v = 0; v < 8; ++v)
            s_hb[wave][(v + g * 8) * D_MODEL + n * 16 + c] = (__bf16)(accH[n][v] + proj_b[n * 16 + c]);
    __syncthreads();

    // ---------------- GEMM2: xz[16x256] = h @ mamba_in_W^T; conv-tap + silu --
    v16bf ha0 = load_a_frag_bf(&s_hb[wave][0], D_MODEL, 0,  lane);
    v16bf ha1 = load_a_frag_bf(&s_hb[wave][0], D_MODEL, 32, lane);
    v8f sz[8];                       // silu(z) tiles kept in registers (C-layout)
#pragma unroll
    for (int t = 0; t < 16; ++t) {
        v8f a = {};
        a = wmma_bf16(ha0, load_b_frag_bf(wmin, D_MODEL, t * 16, 0,  lane), a);
        a = wmma_bf16(ha1, load_b_frag_bf(wmin, D_MODEL, t * 16, 32, lane), a);
        if (t < 8) {                 // xi -> xc = silu(xi * conv_tap3 + conv_b)
            const int d = t * 16 + c;
            const float k3 = conv_W[d * 4 + 3], cb = conv_b[d];
#pragma unroll
            for (int v = 0; v < 8; ++v)
                s_xc[wave][(v + g * 8) * D_INNER + d] = silu_f(a[v] * k3 + cb);
        } else {                     // z -> silu(z)
#pragma unroll
            for (int v = 0; v < 8; ++v) sz[t - 8][v] = silu_f(a[v]);
        }
    }
    __syncthreads();

    // ---------------- GEMM3: dbc[16x48] = xc[16x128] @ xproj^T (padded) ------
    {
        v8f dacc[3] = {};
#pragma unroll
        for (int k = 0; k < 4; ++k) {
            v16bf af = load_a_frag_f32(&s_xc[wave][0], D_INNER, k * 32, lane);
#pragma unroll
            for (int n = 0; n < 3; ++n)
                dacc[n] = wmma_bf16(af, load_b_frag_bf(wxp, D_INNER, n * 16, k * 32, lane), dacc[n]);
        }
#pragma unroll
        for (int n = 0; n < 3; ++n)
#pragma unroll
            for (int v = 0; v < 8; ++v)
                s_dbc[wave][(v + g * 8) * XPROJ_NP + n * 16 + c] = dacc[n][v];
    }
    __syncthreads();

    // ---------------- S[row] = Bm . Cm  (h0==0 collapses the scan) -----------
    if (lane < 16) {
        const float* dr = &s_dbc[wave][lane * XPROJ_NP];
        float s = 0.f;
#pragma unroll
        for (int j = 0; j < D_STATE; ++j) s += dr[DT_RANK + j] * dr[DT_RANK + D_STATE + j];
        s_S[wave][lane] = s;
    }
    __syncthreads();

    // ------ dt = softplus(dbc[:,:4] @ dt_W^T + dt_b); y = (dt*S + D)*xc*silu(z)
#pragma unroll
    for (int t = 0; t < 8; ++t) {
        const int d = t * 16 + c;
        const float w0 = dt_W[d * 4 + 0], w1 = dt_W[d * 4 + 1],
                    w2 = dt_W[d * 4 + 2], w3 = dt_W[d * 4 + 3];
        const float db = dt_b[d], Dp = Dparam[d];
#pragma unroll
        for (int v = 0; v < 8; ++v) {
            const int m = v + g * 8;
            const float* dr = &s_dbc[wave][m * XPROJ_NP];
            const float dtl = db + dr[0] * w0 + dr[1] * w1 + dr[2] * w2 + dr[3] * w3;
            const float dts = softplus_f(dtl);
            const float xcv = s_xc[wave][m * D_INNER + d];
            const float y   = (dts * s_S[wave][m] + Dp) * xcv * sz[t][v];
            s_yb[wave][m * D_INNER + d] = (__bf16)y;
        }
    }
    __syncthreads();

    // ---------------- GEMM4: m[16x64] = y[16x128] @ out_W^T ------------------
    {
        v8f macc[4] = {};
#pragma unroll
        for (int k = 0; k < 4; ++k) {
            v16bf af = load_a_frag_bf(&s_yb[wave][0], D_INNER, k * 32, lane);
#pragma unroll
            for (int n = 0; n < 4; ++n)
                macc[n] = wmma_bf16(af, load_b_frag_bf(wout, D_INNER, n * 16, k * 32, lane), macc[n]);
        }
#pragma unroll
        for (int n = 0; n < 4; ++n)
#pragma unroll
            for (int v = 0; v < 8; ++v)
                s_m[wave][(v + g * 8) * D_MODEL + n * 16 + c] = macc[n][v];
    }
    __syncthreads();

    // ---------------- head: out[16x2] = m @ head_W^T + head_b (scalar) -------
    if (lane < 16) {
        const float* mr = &s_m[wave][lane * D_MODEL];
        float o0 = head_b[0], o1 = head_b[1];
#pragma unroll
        for (int d = 0; d < D_MODEL; ++d) {
            const float mv = mr[d];
            o0 += mv * head_W[d];
            o1 += mv * head_W[D_MODEL + d];
        }
        const size_t row = (size_t)rowBase + lane;
        out[row * 2 + 0] = o0;
        out[row * 2 + 1] = o1;
    }
}

extern "C" void kernel_launch(void* const* d_in, const int* in_sizes, int n_in,
                              void* d_out, int out_size, void* d_ws, size_t ws_size,
                              hipStream_t stream) {
    (void)n_in; (void)out_size; (void)ws_size;
    const float* x       = (const float*)d_in[0];
    const float* proj_W  = (const float*)d_in[1];
    const float* proj_b  = (const float*)d_in[2];
    const float* min_W   = (const float*)d_in[3];
    const float* conv_W  = (const float*)d_in[4];
    const float* conv_b  = (const float*)d_in[5];
    const float* xproj_W = (const float*)d_in[6];
    const float* dt_W    = (const float*)d_in[7];
    const float* dt_b    = (const float*)d_in[8];
    // d_in[9] = A_log: mathematically dead (single scan step from h0 == 0).
    const float* Dparam  = (const float*)d_in[10];
    const float* out_W   = (const float*)d_in[11];
    const float* head_W  = (const float*)d_in[12];
    const float* head_b  = (const float*)d_in[13];
    float* out  = (float*)d_out;
    __bf16* wsb = (__bf16*)d_ws;     // needs WS_TOTAL*2 ~ 160 KB of scratch

    pack_weights<<<64, 256, 0, stream>>>(proj_W, min_W, xproj_W, out_W, wsb);

    const int B = in_sizes[0] / INPUT_DIM;          // 65536
    const int blocks = B / TILE_B;                  // 1024
    mamba_router_fused<<<blocks, WAVES_PER_BLOCK * 32, 0, stream>>>(
        x, wsb, proj_b, conv_W, conv_b, dt_W, dt_b, Dparam, head_W, head_b, out);
}
